// RWKVSelfAttention_83408264888359
// MI455X (gfx1250) — matrix operand order repaired and checked by reading, not compiled, not run on hardware
//
#include <hip/hip_runtime.h>
#include <hip/hip_bf16.h>

#define H 1024
#define B 8
#define T 2048
#define BT (B * T)                   // 16384 rows
#define BTH ((size_t)B * T * H)      // 16,777,216 elements
#define HH ((size_t)H * H)           // 1,048,576 elements

typedef __bf16 bf16;
typedef __attribute__((ext_vector_type(16))) __bf16 v16bf;
typedef __attribute__((ext_vector_type(8)))  __bf16 v8bf;
typedef __attribute__((ext_vector_type(8)))  float  v8f;
typedef __attribute__((ext_vector_type(4)))  int    v4i;

union BF16x16 { v16bf v; v8bf h[2]; };

// gfx1250 async global->LDS copy (ASYNCcnt-tracked), with a synchronous
// register-staging fallback if this toolchain doesn't declare the builtins.
// Probe-confirmed signature: arg0 = AS1 (global) int4*, arg1 = AS3 (LDS)
// int4*, then two int immediates (offset, cpol).
#if defined(__has_builtin)
#if __has_builtin(__builtin_amdgcn_global_load_async_to_lds_b128) && \
    __has_builtin(__builtin_amdgcn_s_wait_asynccnt)
#define USE_ASYNC_LDS 1
#endif
#endif

#define GPTR(p) ((__attribute__((address_space(1))) v4i*)(p))
#define LPTR(p) ((__attribute__((address_space(3))) v4i*)(p))

// ---------------------------------------------------------------------------
// Weight fp32 -> bf16 conversion (4 matrices, selected by blockIdx.y)
// ---------------------------------------------------------------------------
__global__ void cvt_weights(const float* __restrict__ w0, const float* __restrict__ w1,
                            const float* __restrict__ w2, const float* __restrict__ w3,
                            bf16* __restrict__ o0, bf16* __restrict__ o1,
                            bf16* __restrict__ o2, bf16* __restrict__ o3) {
    size_t i = (size_t)blockIdx.x * 256 + threadIdx.x;
    const float* s;
    bf16* d;
    switch (blockIdx.y) {
        case 0:  s = w0; d = o0; break;
        case 1:  s = w1; d = o1; break;
        case 2:  s = w2; d = o2; break;
        default: s = w3; d = o3; break;
    }
    d[i] = (bf16)s[i];
}

// ---------------------------------------------------------------------------
// Time-shift + mix, fp32 -> bf16  (k_in, v_in, r_in)
// ---------------------------------------------------------------------------
__global__ void mix_kernel(const float* __restrict__ x,
                           const float* __restrict__ tmk,
                           const float* __restrict__ tmv,
                           const float* __restrict__ tmr,
                           bf16* __restrict__ kin,
                           bf16* __restrict__ vin,
                           bf16* __restrict__ rin) {
    size_t idx = (size_t)blockIdx.x * 256 + threadIdx.x;     // < BTH
    int h = (int)(idx & (H - 1));
    int t = (int)((idx >> 10) & (T - 1));                    // H == 1024
    float xv = x[idx];
    float xs = (t == 0) ? 0.0f : x[idx - H];
    float a = tmk[h], b = tmv[h], c = tmr[h];
    kin[idx] = (bf16)(xv * a + xs * (1.0f - a));
    vin[idx] = (bf16)(xv * b + xs * (1.0f - b));
    rin[idx] = (bf16)(xv * c + xs * (1.0f - c));
}

// ---------------------------------------------------------------------------
// WMMA GEMM:  C[M, N] = A[M, K] * W[N, K]^T    (M = BT, N = K = H)
// Block = 256 threads (8 waves), 64 rows x 512 cols per block; each wave owns
// a 64x64 tile. The shared 64x32 A tile per K-step is staged once per block
// into LDS (double-buffered, async global->LDS copies when available) and the
// 8 waves read their fragments with conflict-free ds_load_b128 (row stride
// padded to 80B). B fragments stream straight from global (no reuse in-block).
// mode: 0 = plain, 1 = sigmoid(C), 2 = C + bias[col]
// ---------------------------------------------------------------------------
#define LDS_STRIDE 40                 // bf16 elems per LDS row (80 B, padded)

__global__ __launch_bounds__(256)
void wmma_gemm_bf16(const bf16* __restrict__ A, const bf16* __restrict__ W,
                    const float* __restrict__ bias, float* __restrict__ C,
                    int mode) {
    const int K = H, N = H;
    __shared__ bf16 As[2][64 * LDS_STRIDE];   // 2 x 5 KiB

    const int tid  = threadIdx.x;
    const int lane = tid & 31;
    const int wave = tid >> 5;
    const int m    = lane & 15;        // row-in-tile (A) / col-in-tile (B)
    const int sel  = lane >> 4;        // half-wave select
    const int row0 = blockIdx.y * 64;
    const int col0 = blockIdx.x * 512 + wave * 64;

    // Staging map: 256 threads x 16 B cover the 64x32 bf16 (4 KiB) A tile.
    const int srow   = tid >> 2;       // 0..63
    const int schunk = tid & 3;        // 0..3 (8 bf16 each)
    const bf16* gsrc = A + (size_t)(row0 + srow) * K + schunk * 8;
    bf16* lbuf[2] = { &As[0][srow * LDS_STRIDE + schunk * 8],
                      &As[1][srow * LDS_STRIDE + schunk * 8] };

    // B fragment (32x16 bf16): lane holds col n = m, 8 K at kk + sel*8 and
    // 8 K at kk + 16 + sel*8 (matches A); contiguous along K since W is [N, K].
    const bf16* Bbase[4];
#pragma unroll
    for (int i = 0; i < 4; ++i)
        Bbase[i] = W + (size_t)(col0 + i * 16 + m) * K + sel * 8;

    v8f acc[4][4] = {};

    // Prologue: stage K-step 0 into buffer 0.
#if USE_ASYNC_LDS
    __builtin_amdgcn_global_load_async_to_lds_b128(GPTR(gsrc), LPTR(lbuf[0]), 0, 0);
#else
    *(v8bf*)lbuf[0] = *(const v8bf*)gsrc;
#endif

    const int NI = K / 32;
    for (int i = 0; i < NI; ++i) {
        const int kk = i * 32;
#if USE_ASYNC_LDS
        __builtin_amdgcn_s_wait_asynccnt(0);   // own stage(i) complete
#endif
        __syncthreads();                       // stage(i) visible to all waves;
                                               // everyone done reading buf(i+1)
        if (i + 1 < NI) {                      // overlap stage(i+1) w/ compute(i)
            const bf16* gn = gsrc + (i + 1) * 32;
            bf16* ln = lbuf[(i + 1) & 1];
#if USE_ASYNC_LDS
            __builtin_amdgcn_global_load_async_to_lds_b128(GPTR(gn), LPTR(ln), 0, 0);
#else
            *(v8bf*)ln = *(const v8bf*)gn;
#endif
        }

        // A fragment (16x32 bf16, ISA 7.12.2): lane holds 8 halves at
        // K=sel*8 and 8 halves at K=16+sel*8 of row m (from LDS).
        const bf16* abuf = &As[i & 1][0];
        v16bf af[4], bfrag[4];
#pragma unroll
        for (int f = 0; f < 4; ++f) {
            const bf16* ap = abuf + (f * 16 + m) * LDS_STRIDE + sel * 8;
            BF16x16 ua;
            ua.h[0] = *(const v8bf*)(ap);
            ua.h[1] = *(const v8bf*)(ap + 16);
            af[f] = ua.v;
            BF16x16 ub;
            ub.h[0] = *(const v8bf*)(Bbase[f] + kk);
            ub.h[1] = *(const v8bf*)(Bbase[f] + kk + 16);
            bfrag[f] = ub.v;
        }
#pragma unroll
        for (int fi = 0; fi < 4; ++fi)
#pragma unroll
            for (int fj = 0; fj < 4; ++fj)
                acc[fi][fj] = __builtin_amdgcn_wmma_f32_16x16x32_bf16(
                    false, af[fi], false, bfrag[fj], (short)0, acc[fi][fj],
                    false, false);
    }

    // Epilogue. C/D layout: VGPR e of lane -> (M = e + 8*sel, N = lane&15).
#pragma unroll
    for (int i = 0; i < 4; ++i) {
#pragma unroll
        for (int j = 0; j < 4; ++j) {
            const int col = col0 + j * 16 + m;
            const float bv = (mode == 2) ? bias[col] : 0.0f;
#pragma unroll
            for (int e = 0; e < 8; ++e) {
                const int row = row0 + i * 16 + 8 * sel + e;
                float val = acc[i][j][e];
                if (mode == 1) val = 1.0f / (1.0f + __expf(-val));
                else           val += bv;      // bv == 0 for mode 0
                C[(size_t)row * N + col] = val;
            }
        }
    }
}

// ---------------------------------------------------------------------------
// WKV recurrence: one lane per (b, h) channel, serial over t, coalesced over h.
// Emits rwkv = r * wkv as bf16 for the output GEMM.
// ---------------------------------------------------------------------------
__global__ void wkv_scan(const float* __restrict__ k, const float* __restrict__ v,
                         const float* __restrict__ r,
                         const float* __restrict__ time_decay,
                         const float* __restrict__ time_first,
                         bf16* __restrict__ rwkv) {
    int idx = blockIdx.x * 256 + threadIdx.x;   // < B*H
    int h = idx & (H - 1);
    int b = idx >> 10;
    const float td = __expf(-__expf(time_decay[h]));
    const float tf = time_first[h];
    const size_t base = (size_t)b * T * H + h;

    float num = 0.0f, den = 0.0f;
    float kn = k[base], vn = v[base], rn = r[base];
    for (int t = 0; t < T; ++t) {
        const float kt = kn, vt = vn, rt = rn;
        if (t + 1 < T) {                        // prefetch next timestep
            const size_t noff = base + (size_t)(t + 1) * H;
            kn = k[noff]; vn = v[noff]; rn = r[noff];
        }
        const float e   = __expf(tf + kt);
        const float out = (num + e * vt) / (den + e);
        const float ek  = __expf(kt);
        num = td * num + ek * vt;
        den = td * den + ek;
        rwkv[base + (size_t)t * H] = (bf16)(rt * out);
    }
}

// ---------------------------------------------------------------------------
extern "C" void kernel_launch(void* const* d_in, const int* in_sizes, int n_in,
                              void* d_out, int out_size, void* d_ws, size_t ws_size,
                              hipStream_t stream) {
    const float* x          = (const float*)d_in[0];
    const float* time_decay = (const float*)d_in[1];
    const float* time_first = (const float*)d_in[2];
    const float* tmk        = (const float*)d_in[3];
    const float* tmv        = (const float*)d_in[4];
    const float* tmr        = (const float*)d_in[5];
    const float* Wk         = (const float*)d_in[6];
    const float* Wv         = (const float*)d_in[7];
    const float* Wr         = (const float*)d_in[8];
    const float* Wo         = (const float*)d_in[9];
    const float* bo         = (const float*)d_in[10];
    float* out = (float*)d_out;

    char* ws = (char*)d_ws;
    bf16*  kin  = (bf16*)(ws + 0);            // 32 MiB each
    bf16*  vin  = (bf16*)(ws + 33554432);
    bf16*  rin  = (bf16*)(ws + 67108864);
    bf16*  Wkb  = (bf16*)(ws + 100663296);    // 2 MiB each
    bf16*  Wvb  = (bf16*)(ws + 102760448);
    bf16*  Wrb  = (bf16*)(ws + 104857600);
    bf16*  Wob  = (bf16*)(ws + 106954752);
    float* kf   = (float*)(ws + 109051904);   // 64 MiB each
    float* vf   = (float*)(ws + 176160768);
    float* rf   = (float*)(ws + 243269632);
    bf16*  rwkv = kin;                        // kin is dead after GEMM #1

    cvt_weights<<<dim3((unsigned)(HH / 256), 4), 256, 0, stream>>>(
        Wk, Wv, Wr, Wo, Wkb, Wvb, Wrb, Wob);

    mix_kernel<<<(unsigned)(BTH / 256), 256, 0, stream>>>(
        x, tmk, tmv, tmr, kin, vin, rin);

    dim3 g(H / 512, BT / 64);                 // (2, 256)
    wmma_gemm_bf16<<<g, 256, 0, stream>>>(kin, Wkb, nullptr, kf, 0);
    wmma_gemm_bf16<<<g, 256, 0, stream>>>(vin, Wvb, nullptr, vf, 0);
    wmma_gemm_bf16<<<g, 256, 0, stream>>>(rin, Wrb, nullptr, rf, 1);

    wkv_scan<<<(B * H) / 256, 256, 0, stream>>>(
        kf, vf, rf, time_decay, time_first, rwkv);

    wmma_gemm_bf16<<<g, 256, 0, stream>>>(rwkv, Wob, bo, out, 2);
}
